// SelfAttention_22995254903416
// MI455X (gfx1250) — compile-verified, matrix operand order checked by reading, not run on hardware
//
#include <hip/hip_runtime.h>
#include <hip/hip_bf16.h>

// ---------------- problem constants ----------------
#define BATCH   2
#define SEQ     2048
#define DMODEL  1024
#define NH      16
#define DHEAD   64
#define N_QKV   (3 * DMODEL)     // 3072
#define ROWS    (BATCH * SEQ)    // 4096

typedef __bf16 bf16;
typedef __attribute__((ext_vector_type(16))) __bf16 bf16x16;
typedef __attribute__((ext_vector_type(8)))  __bf16 bf16x8;
typedef __attribute__((ext_vector_type(8)))  float  f32x8;
typedef __attribute__((ext_vector_type(4)))  unsigned int u32x4;
typedef __attribute__((ext_vector_type(8)))  int    i32x8;
typedef __attribute__((ext_vector_type(4)))  int    i32x4;

#if defined(__has_builtin)
#if __has_builtin(__builtin_amdgcn_tensor_load_to_lds)
#define HAVE_TDM 1
#endif
#endif

// round-to-nearest-even f32 -> bf16
__device__ __forceinline__ bf16 f2bf(float f) {
    unsigned u = __builtin_bit_cast(unsigned, f);
    u += 0x7FFFu + ((u >> 16) & 1u);
    unsigned short h = (unsigned short)(u >> 16);
    return __builtin_bit_cast(bf16, h);
}

// assemble a 16-element bf16 fragment from two contiguous 16-byte chunks
__device__ __forceinline__ bf16x16 load_frag(const bf16* p0, const bf16* p1) {
    bf16x8 lo = *(const bf16x8*)p0;
    bf16x8 hi = *(const bf16x8*)p1;
    return __builtin_shufflevector(lo, hi, 0,1,2,3,4,5,6,7,8,9,10,11,12,13,14,15);
}

__device__ __forceinline__ f32x8 wmma_bf16(bf16x16 a, bf16x16 b, f32x8 c) {
    return __builtin_amdgcn_wmma_f32_16x16x32_bf16(false, a, false, b, (short)0, c,
                                                   false, false);
}

#ifdef HAVE_TDM
// Issue a 2D TDM load: tile (tile_d1 rows x tile_d0 elems) of 2-byte elements,
// row stride stride0 (elems), from gptr into LDS byte offset lds_off.
// Descriptor layout per CDNA5 ISA sect. 8.
__device__ __forceinline__ void tdm_load_2d(unsigned lds_off, const bf16* gptr,
                                            unsigned tensor_d0, unsigned tensor_d1,
                                            unsigned tile_d0, unsigned tile_d1,
                                            unsigned stride0) {
    unsigned long long ga = (unsigned long long)(__SIZE_TYPE__)gptr;
    u32x4 g0;
    g0[0] = 1u;                                     // count=1 (valid user D#)
    g0[1] = lds_off;                                // lds_addr
    g0[2] = (unsigned)(ga & 0xffffffffu);           // global_addr[31:0]
    g0[3] = (unsigned)((ga >> 32) & 0x1ffffffu) | (2u << 30);  // addr hi | type=2
    i32x8 g1;
    g1[0] = (int)(1u << 16);                        // data_size=1 (2 bytes)
    g1[1] = (int)((tensor_d0 & 0xffffu) << 16);     // tensor_dim0[15:0]
    g1[2] = (int)(((tensor_d0 >> 16) & 0xffffu) |
                  ((tensor_d1 & 0xffffu) << 16));   // dim0 hi | dim1 lo
    g1[3] = (int)(((tensor_d1 >> 16) & 0xffffu) |
                  ((tile_d0 & 0xffffu) << 16));     // dim1 hi | tile_dim0
    g1[4] = (int)(tile_d1 & 0xffffu);               // tile_dim1 (tile_dim2=0)
    g1[5] = (int)stride0;                           // tensor_dim0_stride lo
    g1[6] = 0;
    g1[7] = 0;
    i32x4 z4 = {0, 0, 0, 0};
#if defined(__clang_major__) && __clang_major__ >= 23
    i32x8 z8 = {0, 0, 0, 0, 0, 0, 0, 0};
    __builtin_amdgcn_tensor_load_to_lds(g0, g1, z4, z4, z8, 0);
#else
    __builtin_amdgcn_tensor_load_to_lds(g0, g1, z4, z4, 0);
#endif
}
#endif

// ---------------- conversion kernels ----------------
__global__ void cvt_f32_to_bf16(const float* __restrict__ src,
                                bf16* __restrict__ dst, int n) {
    for (int i = blockIdx.x * blockDim.x + threadIdx.x; i < n;
         i += gridDim.x * blockDim.x)
        dst[i] = f2bf(src[i]);
}

__global__ void transpose_to_bf16(const float* __restrict__ W,
                                  bf16* __restrict__ Wt, int ncols, int n) {
    for (int i = blockIdx.x * blockDim.x + threadIdx.x; i < n;
         i += gridDim.x * blockDim.x) {
        int nn = i >> 10;
        int kk = i & 1023;
        Wt[i] = f2bf(W[kk * ncols + nn]);
    }
}

// ====================================================================
// Tiled GEMM core: each block computes a 64x64 tile of C = A @ B^T
// (B stored K-major).  A/B 64x32 sub-tiles are staged in double-
// buffered LDS by the Tensor Data Mover and consumed by WMMA.
// Load phase and WMMA phase are split so all ds_load_b128 issue
// back-to-back with a single s_wait_dscnt before the WMMA burst.
//
// TDM_WAIT is an asm with a "memory" clobber: it is both the hardware
// s_wait_tensorcnt AND the compiler-level point at which LDS (an
// addrspace(3) global) must be assumed modified -- without it LLVM
// folds the smem loads to undef because the TDM write is invisible.
// ====================================================================
#define GEMM_BODY(APTR, BPTR, NTOT)                                              \
    __shared__ __align__(128) bf16 smem[8192]; /* A:2x2048, B:2x2048 elems */    \
    const int tid  = threadIdx.x;                                                \
    const int lane = tid & 31;                                                   \
    const int wave = tid >> 5;                                                   \
    const int lr   = lane & 15;                                                  \
    const int kb   = (lane >> 4) << 3;                                           \
    const int nbB  = blockIdx.x * 64;                                            \
    const int mbB  = blockIdx.y * 64;                                            \
    const int rowA = wave * 16 + lr;                                             \
    f32x8 acc[4] = {};                                                           \
    stage_tile(smem, tid, wave, 0, 0, APTR, BPTR, mbB, nbB, (NTOT));             \
    TDM_WAIT();                                                                  \
    __syncthreads();                                                             \
    for (int k0 = 0; k0 < DMODEL; k0 += 32) {                                    \
        int cur = (k0 >> 5) & 1;                                                 \
        if (k0 + 32 < DMODEL)                                                    \
            stage_tile(smem, tid, wave, cur ^ 1, k0 + 32, APTR, BPTR, mbB, nbB,  \
                       (NTOT));                                                  \
        const bf16* Al = smem + cur * 2048;                                      \
        const bf16* Bl = smem + 4096 + cur * 2048;                               \
        /* load phase: A frag + all four B frags */                              \
        bf16x16 a = load_frag(Al + rowA * 32 + kb, Al + rowA * 32 + 16 + kb);    \
        bf16x16 bfr[4];                                                          \
        _Pragma("unroll")                                                        \
        for (int i = 0; i < 4; ++i) {                                            \
            const bf16* bp = Bl + (i * 16 + lr) * 32 + kb;                       \
            bfr[i] = load_frag(bp, bp + 16);                                     \
        }                                                                        \
        /* compute phase: WMMA burst */                                          \
        _Pragma("unroll")                                                        \
        for (int i = 0; i < 4; ++i)                                              \
            acc[i] = wmma_bf16(a, bfr[i], acc[i]);                               \
        TDM_WAIT();                                                              \
        __syncthreads();                                                         \
    }

#ifdef HAVE_TDM
#define TDM_WAIT() asm volatile("s_wait_tensorcnt 0" ::: "memory")
__device__ __forceinline__ void stage_tile(bf16* smem, int tid, int wave, int buf,
                                           int k0, const bf16* A, const bf16* B,
                                           int mbB, int nbB, int ntot) {
    if (wave == 0) {
        tdm_load_2d((unsigned)(buf * 4096),
                    A + (__SIZE_TYPE__)mbB * DMODEL + k0,
                    DMODEL, ROWS, 32, 64, DMODEL);
        tdm_load_2d((unsigned)(8192 + buf * 4096),
                    B + (__SIZE_TYPE__)nbB * DMODEL + k0,
                    DMODEL, (unsigned)ntot, 32, 64, DMODEL);
        // escape the LDS base so the memory clobbers above/below alias it
        asm volatile("" : : "s"(smem) : "memory");
    }
}
#else
#define TDM_WAIT() ((void)0)
__device__ __forceinline__ void stage_tile(bf16* smem, int tid, int wave, int buf,
                                           int k0, const bf16* A, const bf16* B,
                                           int mbB, int nbB, int ntot) {
    int row  = tid >> 1;
    int half = (tid & 1) * 16;
    const bf16* gA = A + (__SIZE_TYPE__)(mbB + row) * DMODEL + k0 + half;
    *(bf16x8*)(smem + buf * 2048 + row * 32 + half)     = *(const bf16x8*)gA;
    *(bf16x8*)(smem + buf * 2048 + row * 32 + half + 8) = *(const bf16x8*)(gA + 8);
    const bf16* gB = B + (__SIZE_TYPE__)(nbB + row) * DMODEL + k0 + half;
    *(bf16x8*)(smem + 4096 + buf * 2048 + row * 32 + half)     = *(const bf16x8*)gB;
    *(bf16x8*)(smem + 4096 + buf * 2048 + row * 32 + half + 8) = *(const bf16x8*)(gB + 8);
}
#endif

// ---------------- QKV projection ----------------
__global__ void __launch_bounds__(128)
qkv_gemm(const bf16* __restrict__ xb, const bf16* __restrict__ wT,
         const float* __restrict__ bias,
         bf16* __restrict__ Qb, bf16* __restrict__ Kb, bf16* __restrict__ Vt) {
    GEMM_BODY(xb, wT, N_QKV)
#pragma unroll
    for (int i = 0; i < 4; ++i) {
        int col  = nbB + i * 16 + lr;
        int part = col >> 10;
        int cin  = col & 1023;
        int h    = cin >> 6;
        int d    = cin & 63;
        float bv = bias[col];
#pragma unroll
        for (int j = 0; j < 8; ++j) {
            int row = mbB + wave * 16 + j + kb;
            int b   = row >> 11;
            int q   = row & 2047;
            int bh  = b * NH + h;
            bf16 v  = f2bf(acc[i][j] + bv);
            if (part == 0)      Qb[((__SIZE_TYPE__)(bh * SEQ) + q) * DHEAD + d] = v;
            else if (part == 1) Kb[((__SIZE_TYPE__)(bh * SEQ) + q) * DHEAD + d] = v;
            else                Vt[((__SIZE_TYPE__)(bh * DHEAD) + d) * SEQ + q] = v;
        }
    }
}

// ---------------- output projection + bias + residual ----------------
__global__ void __launch_bounds__(128)
out_gemm(const bf16* __restrict__ ctxb, const bf16* __restrict__ wT,
         const float* __restrict__ bias, const float* __restrict__ x,
         float* __restrict__ out) {
    GEMM_BODY(ctxb, wT, DMODEL)
#pragma unroll
    for (int i = 0; i < 4; ++i) {
        int col  = nbB + i * 16 + lr;
        float bv = bias[col];
#pragma unroll
        for (int j = 0; j < 8; ++j) {
            int row = mbB + wave * 16 + j + kb;
            __SIZE_TYPE__ idx = (__SIZE_TYPE__)row * DMODEL + col;
            out[idx] = acc[i][j] + bv + x[idx];
        }
    }
}

// ---------------- flash attention: one wave per 16-query tile ----------------
__global__ void __launch_bounds__(128)
attn(const bf16* __restrict__ Qb, const bf16* __restrict__ Kb,
     const bf16* __restrict__ Vt, bf16* __restrict__ ctxb) {
    __shared__ __align__(16) bf16 Pbuf[4][16 * 32];

    const int lane  = threadIdx.x & 31;
    const int wave  = threadIdx.x >> 5;
    const int bh    = blockIdx.x >> 5;
    const int q64   = blockIdx.x & 31;
    const int qbase = q64 * 64 + wave * 16;
    const int lr    = lane & 15;
    const int kb    = (lane >> 4) << 3;

    const bf16* Qp = Qb + ((__SIZE_TYPE__)(bh * SEQ) + qbase + lr) * DHEAD;
    bf16x16 qf0 = load_frag(Qp + kb,      Qp + 16 + kb);
    bf16x16 qf1 = load_frag(Qp + 32 + kb, Qp + 48 + kb);

    const bf16* Kbase = Kb + (__SIZE_TYPE__)bh * SEQ * DHEAD;
    const bf16* Vbase = Vt + (__SIZE_TYPE__)bh * DHEAD * SEQ;

    f32x8 o[4] = {};
    float mrow[8], lsum[8];
#pragma unroll
    for (int j = 0; j < 8; ++j) { mrow[j] = -__builtin_inff(); lsum[j] = 0.f; }

    bf16* P = &Pbuf[wave][0];
    const float scale = 0.125f;

    for (int kc = 0; kc < SEQ; kc += 32) {
        if (kc + 32 < SEQ)
            __builtin_prefetch(Kbase + (__SIZE_TYPE__)(kc + 32 + lr) * DHEAD, 0, 0);

        // ---- load phase: all K fragments for the 32-key chunk ----
        const bf16* Kp0 = Kbase + (__SIZE_TYPE__)(kc + lr) * DHEAD;
        const bf16* Kp1 = Kbase + (__SIZE_TYPE__)(kc + 16 + lr) * DHEAD;
        bf16x16 ka0 = load_frag(Kp0 + kb,      Kp0 + 16 + kb);
        bf16x16 ka1 = load_frag(Kp0 + 32 + kb, Kp0 + 48 + kb);
        bf16x16 kb0 = load_frag(Kp1 + kb,      Kp1 + 16 + kb);
        bf16x16 kb1 = load_frag(Kp1 + 32 + kb, Kp1 + 48 + kb);

        // ---- S = Q @ K^T ----
        f32x8 s0 = {}, s1 = {};
        s0 = wmma_bf16(qf0, ka0, s0);
        s0 = wmma_bf16(qf1, ka1, s0);
        s1 = wmma_bf16(qf0, kb0, s1);
        s1 = wmma_bf16(qf1, kb1, s1);

        // ---- hoist V fragment loads: latency hidden by softmax VALU work ----
        bf16x16 vf[4];
#pragma unroll
        for (int n = 0; n < 4; ++n) {
            const bf16* Vp = Vbase + (__SIZE_TYPE__)(n * 16 + lr) * SEQ + kc + kb;
            vf[n] = load_frag(Vp, Vp + 16);
        }

        // ---- online softmax (row reductions via 16-lane butterflies) ----
        float t[8];
#pragma unroll
        for (int j = 0; j < 8; ++j) {
            s0[j] *= scale; s1[j] *= scale;
            t[j] = fmaxf(s0[j], s1[j]);
        }
#pragma unroll
        for (int mask = 1; mask < 16; mask <<= 1)
#pragma unroll
            for (int j = 0; j < 8; ++j)
                t[j] = fmaxf(t[j], __shfl_xor(t[j], mask, 32));

        float corr[8], psum[8];
#pragma unroll
        for (int j = 0; j < 8; ++j) {
            float mn = fmaxf(mrow[j], t[j]);
            corr[j]  = __expf(mrow[j] - mn);
            mrow[j]  = mn;
            s0[j]    = __expf(s0[j] - mn);
            s1[j]    = __expf(s1[j] - mn);
            psum[j]  = s0[j] + s1[j];
        }
#pragma unroll
        for (int mask = 1; mask < 16; mask <<= 1)
#pragma unroll
            for (int j = 0; j < 8; ++j)
                psum[j] += __shfl_xor(psum[j], mask, 32);
#pragma unroll
        for (int j = 0; j < 8; ++j) {
            lsum[j] = lsum[j] * corr[j] + psum[j];
            o[0][j] *= corr[j]; o[1][j] *= corr[j];
            o[2][j] *= corr[j]; o[3][j] *= corr[j];
        }

        // ---- P tile -> LDS (layout transpose for A-fragment) ----
#pragma unroll
        for (int j = 0; j < 8; ++j) {
            P[(j + kb) * 32 + lr]      = f2bf(s0[j]);
            P[(j + kb) * 32 + 16 + lr] = f2bf(s1[j]);
        }
        asm volatile("s_wait_dscnt 0" ::: "memory");
        bf16x16 pa = load_frag(P + lr * 32 + kb, P + lr * 32 + 16 + kb);

        // ---- O += P @ V ----
#pragma unroll
        for (int n = 0; n < 4; ++n)
            o[n] = wmma_bf16(pa, vf[n], o[n]);
    }

    const int h = bh & (NH - 1);
    const int b = bh >> 4;
#pragma unroll
    for (int n = 0; n < 4; ++n) {
        int col = h * DHEAD + n * 16 + lr;
#pragma unroll
        for (int j = 0; j < 8; ++j) {
            int q = qbase + j + kb;
            ctxb[((__SIZE_TYPE__)(b * SEQ) + q) * DMODEL + col] = f2bf(o[n][j] / lsum[j]);
        }
    }
}

// ---------------- host launcher ----------------
extern "C" void kernel_launch(void* const* d_in, const int* in_sizes, int n_in,
                              void* d_out, int out_size, void* d_ws, size_t ws_size,
                              hipStream_t stream) {
    const float* x    = (const float*)d_in[0];
    const float* Wqkv = (const float*)d_in[1];
    const float* bqkv = (const float*)d_in[2];
    const float* Wout = (const float*)d_in[3];
    const float* bout = (const float*)d_in[4];
    float* out = (float*)d_out;

    char* ws = (char*)d_ws;
    const size_t MB = 1024 * 1024;
    bf16* xb    = (bf16*)(ws + 0);
    bf16* wqkvT = (bf16*)(ws + 8  * MB);
    bf16* woutT = (bf16*)(ws + 14 * MB);
    bf16* Qb    = (bf16*)(ws + 16 * MB);
    bf16* Kb    = (bf16*)(ws + 24 * MB);
    bf16* Vt    = (bf16*)(ws + 32 * MB);
    bf16* ctxb  = (bf16*)(ws + 40 * MB);

    cvt_f32_to_bf16<<<4096, 256, 0, stream>>>(x, xb, ROWS * DMODEL);
    transpose_to_bf16<<<12288, 256, 0, stream>>>(Wqkv, wqkvT, N_QKV, N_QKV * DMODEL);
    transpose_to_bf16<<<4096, 256, 0, stream>>>(Wout, woutT, DMODEL, DMODEL * DMODEL);

    qkv_gemm<<<dim3(N_QKV / 64, ROWS / 64), 128, 0, stream>>>(xb, wqkvT, bqkv,
                                                              Qb, Kb, Vt);

    attn<<<BATCH * NH * (SEQ / 64), 128, 0, stream>>>(Qb, Kb, Vt, ctxb);

    out_gemm<<<dim3(DMODEL / 64, ROWS / 64), 128, 0, stream>>>(ctxb, woutT, bout,
                                                               x, out);
}
